// CrossMambaBlock_15977278341331
// MI455X (gfx1250) — matrix operand order repaired and checked by reading, not compiled, not run on hardware
//
#include <hip/hip_runtime.h>
#include <stdint.h>
#include <math.h>

typedef __attribute__((ext_vector_type(16))) _Float16 v16h;
typedef __attribute__((ext_vector_type(8)))  float    v8f;
typedef __attribute__((ext_vector_type(4)))  unsigned u32x4;
typedef __attribute__((ext_vector_type(8)))  unsigned u32x8;

constexpr int Bsz   = 4;
constexpr int Nseq  = 1024;
constexpr int Cdim  = 256;
constexpr int DM    = 512;   // D_MODEL
constexpr int DI    = 512;   // D_INNER
constexpr int DS    = 16;    // D_STATE
constexpr int DTR   = 32;    // DT_RANK
constexpr int Mrows = Bsz * Nseq; // 4096

union Frag { v16h v; uint4 q[2]; };

static __device__ __forceinline__ float nanfix(float x) {
  if (x != x) return 0.f;
  if (x == INFINITY) return 1.f;
  if (x == -INFINITY) return -1.f;
  return x;
}

// wave32 + LDS block reduction (blockDim.x multiple of 32, <=256)
static __device__ __forceinline__ float block_sum(float v, float* sm) {
#pragma unroll
  for (int o = 16; o > 0; o >>= 1) v += __shfl_xor(v, o, 32);
  __syncthreads();
  if ((threadIdx.x & 31) == 0) sm[threadIdx.x >> 5] = v;
  __syncthreads();
  float r = 0.f;
  int nw = blockDim.x >> 5;
  for (int i = 0; i < nw; i++) r += sm[i];
  return r;
}

// ---- Tensor Data Mover: DMA a (rows x K) fp16 panel (row stride ldw) into LDS ----
// D# per CDNA5 ISA ch.8: group0 {count, lds_addr, global_addr, type=2},
// group1 {data_size=2B, tensor_dim0=K, tensor_dim1=rows, tile_dim0=K, tile_dim1=rows,
//         tensor_dim0_stride=ldw}. 2D tensor -> only groups 0/1 (VADDR2/3 NULL).
static __device__ __forceinline__ void tdm_load_b_panel(
    const _Float16* gsrc, unsigned ldsOff, int rows, int K, int ldw) {
  unsigned long long ga = (unsigned long long)(uintptr_t)gsrc;
  u32x4 g0;
  g0[0] = 1u;                                                  // count=1 (valid user D#)
  g0[1] = ldsOff;                                              // lds_addr (bytes)
  g0[2] = (unsigned)ga;                                        // global_addr[31:0]
  g0[3] = (unsigned)((ga >> 32) & 0x01FFFFFFu) | (2u << 30);   // addr[56:32] | type=2
  u32x8 g1;
  g1[0] = 0x00010000u;                                         // data_size=1 -> 2 bytes
  g1[1] = ((unsigned)K & 0xFFFFu) << 16;                       // tensor_dim0 lo16
  g1[2] = ((unsigned)K >> 16) | (((unsigned)rows & 0xFFFFu) << 16); // td0 hi | td1 lo
  g1[3] = (((unsigned)rows >> 16) & 0xFFFFu) | (((unsigned)K & 0xFFFFu) << 16); // td1 hi | tile_dim0
  g1[4] = (unsigned)rows & 0xFFFFu;                            // tile_dim1 | tile_dim2=0
  g1[5] = (unsigned)ldw;                                       // tensor_dim0_stride lo32
  g1[6] = 0u;
  g1[7] = 0u;
  asm volatile("tensor_load_to_lds %0, %1" :: "s"(g0), "s"(g1) : "memory");
}

// ---------------- fp32 -> fp16 weight cast ----------------
__global__ void k_cast_f16(const float* __restrict__ src, _Float16* __restrict__ dst, int n) {
  int i = blockIdx.x * blockDim.x + threadIdx.x;
  if (i < n) dst[i] = (_Float16)src[i];
}

// ---------------- nan_to_num + dual LayerNorm + pack combined ----------------
__global__ __launch_bounds__(256) void k_prep(
    const float* __restrict__ i0, const float* __restrict__ i1,
    const float* __restrict__ g0, const float* __restrict__ b0,
    const float* __restrict__ g1, const float* __restrict__ b1,
    float* __restrict__ x0n, _Float16* __restrict__ comb) {
  __shared__ float sm[8];
  int row = blockIdx.x, t = threadIdx.x;
  float a = nanfix(i0[(size_t)row * Cdim + t]);
  float c = nanfix(i1[(size_t)row * Cdim + t]);
  float s0 = block_sum(a, sm);
  float q0 = block_sum(a * a, sm);
  float s1 = block_sum(c, sm);
  float q1 = block_sum(c * c, sm);
  const float inv = 1.f / (float)Cdim;
  float m0 = s0 * inv, v0 = q0 * inv - m0 * m0;
  float m1 = s1 * inv, v1 = q1 * inv - m1 * m1;
  float xa = (a - m0) * rsqrtf(v0 + 1e-5f) * g0[t] + b0[t];
  float xc = (c - m1) * rsqrtf(v1 + 1e-5f) * g1[t] + b1[t];
  x0n[(size_t)row * Cdim + t] = xa;
  comb[(size_t)row * DM + t] = (_Float16)xa;
  comb[(size_t)row * DM + Cdim + t] = (_Float16)xc;
}

// ---------------- WMMA f16 GEMM with TDM-staged weight panel ----------------
// C = A(MxK) * W(NcxK)^T. Block = 8 waves (4 along M x 2 along N): tile 256x64.
// Grid: (M/256, Nc/64). W panel (64 x K) DMA'd into LDS by the TDM, B fragments
// read back via ds_load_b128; A fragments stream from global (L2-resident).
__global__ __launch_bounds__(256) void k_gemm_tdm(
    const _Float16* __restrict__ A, int lda,
    const _Float16* __restrict__ W, int ldw,
    float* __restrict__ Co, _Float16* __restrict__ Ch,
    const float* __restrict__ bias,
    int Nc, int K, int storeBEL, int epiSoftplus) {
  __shared__ _Float16 smem[64 * 512];   // max panel: 64 rows x K=512 halves = 64KB
  const int wave = threadIdx.x >> 5;
  const int lane = threadIdx.x & 31;
  const int l15 = lane & 15;
  const int hi = lane >> 4;
  const int wm = wave & 3;
  const int wn = wave >> 2;
  const int m0 = blockIdx.x * 256 + wm * 64;
  const int n0blk = blockIdx.y * 64;
  const int n0 = n0blk + wn * 32;

  if (threadIdx.x < 32) {   // one wave issues the DMA; EXEC stays full inside
    tdm_load_b_panel(W + (size_t)n0blk * ldw, (unsigned)(uintptr_t)smem, 64, K, ldw);
    __builtin_amdgcn_s_wait_tensorcnt(0);
  }
  __syncthreads();

  // A 16x32 f16 layout: lanes 0-15 hold K {0..7,16..23}; lanes 16-31 hold K {8..15,24..31}
  const _Float16* ap0 = A + (size_t)(m0 + l15) * lda + hi * 8;
  // B 32x16 f16 layout: lanes 0-15 hold K {0..7,16..23}; lanes 16-31 hold K {8..15,24..31} (col = lane&15)
  const _Float16* bsm = smem + (size_t)(wn * 32 + l15) * K + hi * 8;

  v8f acc[8] = {};
  for (int k0 = 0; k0 < K; k0 += 32) {
    Frag a[4], bf[2];
#pragma unroll
    for (int i = 0; i < 4; i++) {
      const _Float16* p = ap0 + (size_t)(i * 16) * lda + k0;
      a[i].q[0] = *(const uint4*)(p);
      a[i].q[1] = *(const uint4*)(p + 16);
    }
    bf[0].q[0] = *(const uint4*)(bsm + k0);
    bf[0].q[1] = *(const uint4*)(bsm + k0 + 16);
    bf[1].q[0] = *(const uint4*)(bsm + (size_t)16 * K + k0);
    bf[1].q[1] = *(const uint4*)(bsm + (size_t)16 * K + k0 + 16);
    if (k0 + 32 < K) {
      __builtin_prefetch(ap0 + k0 + 32, 0, 3);
      __builtin_prefetch(ap0 + (size_t)32 * lda + k0 + 32, 0, 3);
    }
#pragma unroll
    for (int i = 0; i < 4; i++) {
      acc[i * 2 + 0] = __builtin_amdgcn_wmma_f32_16x16x32_f16(false, a[i].v, false, bf[0].v, (short)0, acc[i * 2 + 0], false, false);
      acc[i * 2 + 1] = __builtin_amdgcn_wmma_f32_16x16x32_f16(false, a[i].v, false, bf[1].v, (short)0, acc[i * 2 + 1], false, false);
    }
  }

  // C/D 16x16 f32 layout: lane L holds col (L&15), rows (L>>4)*8 + e
  const int rb = hi * 8;
#pragma unroll
  for (int ti = 0; ti < 4; ti++) {
#pragma unroll
    for (int tj = 0; tj < 2; tj++) {
      v8f av = acc[ti * 2 + tj];
      int col = n0 + tj * 16 + l15;
      float bval = bias ? bias[col] : 0.f;
#pragma unroll
      for (int e = 0; e < 8; e++) {
        int row = m0 + ti * 16 + rb + e;
        float v = av[e] + bval;
        if (epiSoftplus) v = (v > 20.f) ? v : log1pf(__expf(v));
        size_t idx = storeBEL
            ? ((((size_t)(row >> 10) * Nc + col) << 10) + (size_t)(row & 1023))
            : ((size_t)row * Nc + col);
        Co[idx] = v;
        if (Ch) Ch[(size_t)row * Nc + col] = (_Float16)v;
      }
    }
  }
}

// ---------------- LN + sigmoid + clip (gate weights), in place ----------------
__global__ __launch_bounds__(256) void k_weight(
    float* __restrict__ cwout, const float* __restrict__ g, const float* __restrict__ b) {
  __shared__ float sm[8];
  int row = blockIdx.x, t = threadIdx.x;
  float v = cwout[(size_t)row * Cdim + t];
  float s = block_sum(v, sm);
  float q = block_sum(v * v, sm);
  const float inv = 1.f / (float)Cdim;
  float m = s * inv, var = q * inv - m * m;
  float x = (v - m) * rsqrtf(var + 1e-5f) * g[t] + b[t];
  float w = 1.f / (1.f + __expf(-x));
  w = fminf(0.99f, fmaxf(0.01f, w));
  cwout[(size_t)row * Cdim + t] = w;
}

// ---------------- depthwise causal conv (taps=4) + SiLU; writes (b,l,d) f32+f16 ----------------
__global__ __launch_bounds__(256) void k_conv(
    const float* __restrict__ xz, const float* __restrict__ cw,
    const float* __restrict__ cb, float* __restrict__ xcf,
    _Float16* __restrict__ xch, int flip) {
  int tid = blockIdx.x * blockDim.x + threadIdx.x;
  if (tid >= Bsz * Nseq * DI) return;
  int d = tid & (DI - 1);
  int l = (tid >> 9) & (Nseq - 1);
  int b = tid >> 19;
  float acc = cb[d];
  const float* xrow = xz + ((size_t)b * (2 * DI) + d) * Nseq;
#pragma unroll
  for (int j = 0; j < 4; j++) {
    int lj = l - 3 + j;
    if (lj >= 0) {
      int src = flip ? (Nseq - 1 - lj) : lj;
      acc += cw[d * 4 + j] * xrow[src];
    }
  }
  float s = acc / (1.f + __expf(-acc));
  xcf[tid] = s;
  xch[tid] = (_Float16)s;
}

// ---------------- fused dual-direction selective scan ----------------
// grid = 2*B blocks (dir-major), 512 threads = channel d. B/C broadcast per wave
// via __shfl (lane i of each wave loads one of the 32 contiguous B/C floats) ->
// no LDS, no barriers; waves fully independent on the latency-bound recurrence.
__global__ __launch_bounds__(512) void k_scan(
    const float* __restrict__ delta0, const float* __restrict__ u0,
    const float* __restrict__ xdbl0, size_t strideD, size_t strideX,
    const float* __restrict__ AlogF, const float* __restrict__ AlogB,
    const float* __restrict__ DvF, const float* __restrict__ DvB,
    const float* __restrict__ xz, float* __restrict__ yF, float* __restrict__ yB) {
  const int dir = blockIdx.x >> 2;         // 0 = forward, 1 = backward
  const int b = blockIdx.x & (Bsz - 1);
  const int d = threadIdx.x;
  const int lane = d & 31;
  const float* delta = delta0 + (size_t)dir * strideD;
  const float* u     = u0     + (size_t)dir * strideD;
  const float* xdbl  = xdbl0  + (size_t)dir * strideX;
  const float* Alog  = dir ? AlogB : AlogF;
  const float* Dv    = dir ? DvB : DvF;
  float* yout        = dir ? yB : yF;

  float Ar[DS], h[DS];
#pragma unroll
  for (int n = 0; n < DS; n++) { Ar[n] = -__expf(Alog[d * DS + n]); h[n] = 0.f; }
  float Dd = Dv[d];
  const float* zrow = xz + ((size_t)b * (2 * DI) + DI + d) * Nseq;

  for (int t = 0; t < Nseq; t++) {
    size_t r = (size_t)b * Nseq + t;
    float bc = xdbl[r * 64 + DTR + lane];          // lanes 0..15 -> B, 16..31 -> C
    float dl = delta[r * DI + d];
    float uu = u[r * DI + d];
    float du = dl * uu;
    float y = 0.f;
#pragma unroll
    for (int n = 0; n < DS; n++) {
      float dA = __expf(dl * Ar[n]);
      h[n] = dA * h[n] + du * __shfl(bc, n, 32);
      y += h[n] * __shfl(bc, DS + n, 32);
    }
    y += Dd * uu;
    int src = dir ? (Nseq - 1 - t) : t;            // z gate index AND un-flipped output pos
    float z = zrow[src];
    y *= z / (1.f + __expf(-z));
    yout[((size_t)b * Nseq + src) * DI + d] = y;
  }
}

// ---------------- (yf+yb)*0.5 then LN over 512 -> f16 ----------------
__global__ __launch_bounds__(256) void k_mnorm(
    const float* __restrict__ yf, const float* __restrict__ yb,
    const float* __restrict__ g, const float* __restrict__ bb,
    _Float16* __restrict__ out) {
  __shared__ float sm[8];
  int row = blockIdx.x, t = threadIdx.x;
  size_t base = (size_t)row * DI;
  float v0 = (yf[base + t] + yb[base + t]) * 0.5f;
  float v1 = (yf[base + t + 256] + yb[base + t + 256]) * 0.5f;
  float s = block_sum(v0 + v1, sm);
  float q = block_sum(v0 * v0 + v1 * v1, sm);
  const float inv = 1.f / (float)DI;
  float m = s * inv, var = q * inv - m * m;
  float rs = rsqrtf(var + 1e-5f);
  out[base + t]       = (_Float16)((v0 - m) * rs * g[t] + bb[t]);
  out[base + t + 256] = (_Float16)((v1 - m) * rs * g[t + 256] + bb[t + 256]);
}

// ---------------- LN over 512 of f32 input -> f16 ----------------
__global__ __launch_bounds__(256) void k_pnorm(
    const float* __restrict__ x, const float* __restrict__ g,
    const float* __restrict__ bb, _Float16* __restrict__ out) {
  __shared__ float sm[8];
  int row = blockIdx.x, t = threadIdx.x;
  size_t base = (size_t)row * DM;
  float v0 = x[base + t];
  float v1 = x[base + t + 256];
  float s = block_sum(v0 + v1, sm);
  float q = block_sum(v0 * v0 + v1 * v1, sm);
  const float inv = 1.f / (float)DM;
  float m = s * inv, var = q * inv - m * m;
  float rs = rsqrtf(var + 1e-5f);
  out[base + t]       = (_Float16)((v0 - m) * rs * g[t] + bb[t]);
  out[base + t + 256] = (_Float16)((v1 - m) * rs * g[t + 256] + bb[t + 256]);
}

// ---------------- final blend + skip ----------------
__global__ __launch_bounds__(256) void k_final(
    const float* __restrict__ out2, const float* __restrict__ weight,
    const float* __restrict__ x0n, const float* __restrict__ i0,
    float* __restrict__ out) {
  size_t idx = (size_t)blockIdx.x * blockDim.x + threadIdx.x;
  if (idx >= (size_t)Mrows * Cdim) return;
  float w = weight[idx];
  float v = nanfix(out2[idx]);
  out[idx] = v * w + x0n[idx] * (1.f - w) + nanfix(i0[idx]);
}

extern "C" void kernel_launch(void* const* d_in, const int* in_sizes, int n_in,
                              void* d_out, int out_size, void* d_ws, size_t ws_size,
                              hipStream_t stream) {
  (void)in_sizes; (void)n_in; (void)out_size; (void)ws_size;
  const float* in0 = (const float*)d_in[0];

  // ---- carve workspace ----
  char* ws = (char*)d_ws;
  size_t off = 0;
  auto carve = [&](size_t bytes) -> void* {
    void* p = (void*)(ws + off);
    off += (bytes + 255) & ~(size_t)255;
    return p;
  };
  const size_t strideD = (size_t)Mrows * DI;   // per-direction (b,l,d) f32 elems
  const size_t strideX = (size_t)Mrows * 64;   // per-direction x_dbl elems

  _Float16* cw_h       = (_Float16*)carve((size_t)Cdim * DM * 2);
  _Float16* inproj_h   = (_Float16*)carve((size_t)2 * DI * DM * 2);
  _Float16* xprojf_h   = (_Float16*)carve((size_t)64 * DI * 2);
  _Float16* xprojb_h   = (_Float16*)carve((size_t)64 * DI * 2);
  _Float16* dtf_h      = (_Float16*)carve((size_t)DI * DTR * 2);
  _Float16* dtb_h      = (_Float16*)carve((size_t)DI * DTR * 2);
  _Float16* outproj_h  = (_Float16*)carve((size_t)DM * DI * 2);
  _Float16* projback_h = (_Float16*)carve((size_t)Cdim * DM * 2);
  _Float16* comb_h     = (_Float16*)carve((size_t)Mrows * DM * 2);
  float*    x0n        = (float*)carve((size_t)Mrows * Cdim * 4);
  float*    weight     = (float*)carve((size_t)Mrows * Cdim * 4);
  float*    xzbuf      = (float*)carve((size_t)Bsz * 2 * DI * Nseq * 4);
  float*    xconv_f    = (float*)carve(2 * strideD * 4);
  _Float16* xconv_h    = (_Float16*)carve(2 * strideD * 2);
  float*    xdbl_f     = (float*)carve(2 * strideX * 4);
  _Float16* xdbl_h     = (_Float16*)carve(2 * strideX * 2);
  float*    delta      = (float*)carve(2 * strideD * 4);
  float*    yf         = (float*)carve(strideD * 4);
  float*    yb         = (float*)carve(strideD * 4);
  _Float16* ynorm_h    = (_Float16*)carve((size_t)Mrows * DI * 2);
  float*    out1       = (float*)carve((size_t)Mrows * DM * 4);
  _Float16* pn_h       = (_Float16*)carve((size_t)Mrows * DM * 2);
  float*    out2       = (float*)carve((size_t)Mrows * Cdim * 4);

  auto cast = [&](const float* s, _Float16* d, int n) {
    k_cast_f16<<<(n + 255) / 256, 256, 0, stream>>>(s, d, n);
  };
  cast((const float*)d_in[28], cw_h,       Cdim * DM);     // cw_w
  cast((const float*)d_in[6],  inproj_h,   2 * DI * DM);   // in_proj_w
  cast((const float*)d_in[9],  xprojf_h,   64 * DI);       // xprojf_w
  cast((const float*)d_in[16], xprojb_h,   64 * DI);       // xprojb_w
  cast((const float*)d_in[10], dtf_h,      DI * DTR);      // dtf_w
  cast((const float*)d_in[17], dtb_h,      DI * DTR);      // dtb_w
  cast((const float*)d_in[23], outproj_h,  DM * DI);       // outproj_w
  cast((const float*)d_in[26], projback_h, Cdim * DM);     // projback_w

  k_prep<<<Mrows, 256, 0, stream>>>(in0, (const float*)d_in[1],
      (const float*)d_in[2], (const float*)d_in[3],
      (const float*)d_in[4], (const float*)d_in[5], x0n, comb_h);

  auto gemm = [&](const _Float16* A, int lda, const _Float16* W, int ldw,
                  float* Co, _Float16* Ch, const float* bias,
                  int Nc, int K, int bel, int sp) {
    dim3 g(Mrows / 256, Nc / 64);
    k_gemm_tdm<<<g, 256, 0, stream>>>(A, lda, W, ldw, Co, Ch, bias, Nc, K, bel, sp);
  };

  // gate-weight branch: combined @ cw_w^T + cw_b, then LN/sigmoid/clip in place
  gemm(comb_h, DM, cw_h, DM, weight, nullptr, (const float*)d_in[29], Cdim, DM, 0, 0);
  k_weight<<<Mrows, 256, 0, stream>>>(weight, (const float*)d_in[30], (const float*)d_in[31]);

  // xz = combined @ in_proj_w^T, stored transposed (b, e, l)
  gemm(comb_h, DM, inproj_h, DM, xzbuf, nullptr, nullptr, 2 * DI, DM, 1, 0);

  for (int dir = 0; dir < 2; dir++) {
    const float*    convw  = (const float*)d_in[dir ? 14 : 7];
    const float*    convb  = (const float*)d_in[dir ? 15 : 8];
    const _Float16* xprojh = dir ? xprojb_h : xprojf_h;
    const _Float16* dth    = dir ? dtb_h : dtf_h;
    const float*    dtbias = (const float*)d_in[dir ? 18 : 11];

    float*    xcf = xconv_f + dir * strideD;
    _Float16* xch = xconv_h + dir * strideD;
    float*    xdf = xdbl_f + dir * strideX;
    _Float16* xdh = xdbl_h + dir * strideX;
    float*    dlt = delta + dir * strideD;

    int nconv = Bsz * Nseq * DI;
    k_conv<<<(nconv + 255) / 256, 256, 0, stream>>>(xzbuf, convw, convb, xcf, xch, dir);
    // x_dbl = xconv @ xproj_w^T (f32 + f16 copy for the delta GEMM)
    gemm(xch, DI, xprojh, DI, xdf, xdh, nullptr, 64, DI, 0, 0);
    // delta = softplus(dt @ dt_w^T + dt_b), K=32 (single WMMA k-step)
    gemm(xdh, 64, dth, DTR, dlt, nullptr, dtbias, DI, DTR, 0, 1);
  }

  // both directions scanned concurrently (8 blocks)
  k_scan<<<2 * Bsz, DI, 0, stream>>>(delta, xconv_f, xdbl_f, strideD, strideX,
      (const float*)d_in[12], (const float*)d_in[19],
      (const float*)d_in[13], (const float*)d_in[20],
      xzbuf, yf, yb);

  k_mnorm<<<Mrows, 256, 0, stream>>>(yf, yb, (const float*)d_in[21], (const float*)d_in[22], ynorm_h);
  gemm(ynorm_h, DI, outproj_h, DI, out1, nullptr, nullptr, DM, DI, 0, 0);
  k_pnorm<<<Mrows, 256, 0, stream>>>(out1, (const float*)d_in[24], (const float*)d_in[25], pn_h);
  gemm(pn_h, DM, projback_h, DM, out2, nullptr, (const float*)d_in[27], Cdim, DM, 0, 0);
  k_final<<<(Mrows * Cdim + 255) / 256, 256, 0, stream>>>(out2, weight, x0n, in0, (float*)d_out);
}
